// BoeNet_34574486733234
// MI455X (gfx1250) — matrix operand order repaired and checked by reading, not compiled, not run on hardware
//
#include <hip/hip_runtime.h>
#include <hip/hip_bf16.h>

// ---------------------------------------------------------------------------
// BoeNet forward on gfx1250 (MI455X), wave32.
//
// Dominant cost: logits = pooled(4096x256) @ out_W^T(256x50257)  ~105 GFLOP.
//  - bf16 split-accumulation (a_hi+a_lo)x(w_hi+w_lo), dropping lo*lo:
//    ~2^-18 relative error, 8/3x the fp32-matrix rate via
//    V_WMMA_F32_16X16X32_BF16.
//  - out_W hi/lo (51.5MB total) stays resident in the 192MB L2; logits
//    (823MB) streamed out with non-temporal stores (HBM floor ~35us).
// Stage-A (proj / policy / child GEMMs, ~3.8 GFLOP) stays on full-fp32
// V_WMMA_F32_16X16X4_F32 because the grow decisions are sign thresholds.
// ---------------------------------------------------------------------------

typedef float v2f __attribute__((ext_vector_type(2)));
typedef float v8f __attribute__((ext_vector_type(8)));
typedef __bf16 v16bf __attribute__((ext_vector_type(16)));
typedef unsigned int u32x4 __attribute__((ext_vector_type(4)));
typedef unsigned short ushort_t;

#define HID 256
#define NTOK 4096
#define VOCAB 50257
#define SIB_SCALE 0.0625f /* 1/sqrt(256) */

// ---- bf16 split helpers (round-to-nearest-even) ---------------------------
__device__ __forceinline__ ushort_t bf16_rne(float x) {
  unsigned int u = __float_as_uint(x);
  u += 0x7FFFu + ((u >> 16) & 1u);
  return (ushort_t)(u >> 16);
}
__device__ __forceinline__ float bf16_to_f32(ushort_t h) {
  return __uint_as_float(((unsigned int)h) << 16);
}
__device__ __forceinline__ void split_bf16(float x, ushort_t& hi, ushort_t& lo) {
  hi = bf16_rne(x);
  lo = bf16_rne(x - bf16_to_f32(hi));
}

__device__ __forceinline__ v16bf load_frag16(const ushort_t* p0,
                                             const ushort_t* p1) {
  union {
    v16bf v;
    u32x4 q[2];
  } f;
  f.q[0] = *(const u32x4*)p0;
  f.q[1] = *(const u32x4*)p1;
  return f.v;
}

// ---------------------------------------------------------------------------
// Stage-A GEMM (full fp32): Cout[m][n] = act(sum_k A[m][k]W[n][k] + b1 (+b2))
// A row-major lda=256, optional row-gather. W row-major (Ncols x 256).
// Wave: 64(M) x 16(N) tile via 4 accs; block = 8 waves = 64 x 128.
template <int ACT, bool GATHER>
__global__ __launch_bounds__(256) void gemm_nt_wmma(
    const float* __restrict__ A, const int* __restrict__ gidx,
    const float* __restrict__ W, const float* __restrict__ bias1,
    const float* __restrict__ bias2, float* __restrict__ Cout, int Ncols) {
  const int lane = threadIdx.x & 31;
  const int wave = threadIdx.x >> 5;
  const int m0 = blockIdx.y * 64;
  const int n0 = (blockIdx.x * 8 + wave) * 16;
  const int r = lane & 15;
  const int kk = (lane >> 4) * 2;

  const float* arow[4];
#pragma unroll
  for (int i = 0; i < 4; ++i) {
    int row = m0 + i * 16 + r;
    int src = GATHER ? gidx[row] : row;
    arow[i] = A + (size_t)src * HID + kk;
  }
  int wr = n0 + r;
  if (wr > Ncols - 1) wr = Ncols - 1;
  const float* wrow = W + (size_t)wr * HID + kk;

  v8f acc[4] = {};
#pragma unroll 4
  for (int kt = 0; kt < HID / 4; ++kt) {
    const int k = kt * 4;
    v2f b = *(const v2f*)(wrow + k);
#pragma unroll
    for (int i = 0; i < 4; ++i) {
      v2f a = *(const v2f*)(arow[i] + k);
      acc[i] = __builtin_amdgcn_wmma_f32_16x16x4_f32(false, a, false, b,
                                                     (short)0, acc[i], false,
                                                     false);
    }
  }

  const int col = n0 + r;
  int bcol = col < Ncols ? col : Ncols - 1;
  float bv = 0.0f;
  if (bias1) bv += bias1[bcol];
  if (bias2) bv += bias2[bcol];

  if (col < Ncols) {
    const int mbase = m0 + ((lane >> 4) * 8);
#pragma unroll
    for (int i = 0; i < 4; ++i) {
#pragma unroll
      for (int v = 0; v < 8; ++v) {
        float x = acc[i][v] + bv;
        if (ACT == 1) x = tanhf(x);
        Cout[(size_t)(mbase + i * 16 + v) * (size_t)Ncols + col] = x;
      }
    }
  }
}

// ---------------------------------------------------------------------------
// Logits GEMM: bf16 split (hi*hi + hi*lo + lo*hi) on V_WMMA_F32_16X16X32_BF16.
// A hi/lo: row-major bf16 (M x 256).  W hi/lo: row-major bf16 (Ncols x 256).
// ISA layouts: A(16x32) lane<16 holds K{0-7,16-23}, lane>=16 K{8-15,24-31};
//              B(32x16) lane<16 holds K{0-7,16-23}, lane>=16 K{8-15,24-31}.
__global__ __launch_bounds__(256) void gemm_bf16x3(
    const ushort_t* __restrict__ Ahi, const ushort_t* __restrict__ Alo,
    const ushort_t* __restrict__ Whi, const ushort_t* __restrict__ Wlo,
    const float* __restrict__ bias, float* __restrict__ Cout, int Ncols) {
  const int lane = threadIdx.x & 31;
  const int wave = threadIdx.x >> 5;
  const int m0 = blockIdx.y * 64;
  const int n0 = (blockIdx.x * 8 + wave) * 16;
  const int r = lane & 15;
  const int kh = lane >> 4;

  size_t aoff[4];
#pragma unroll
  for (int i = 0; i < 4; ++i)
    aoff[i] = (size_t)(m0 + i * 16 + r) * HID + kh * 8;
  int wr = n0 + r;
  if (wr > Ncols - 1) wr = Ncols - 1;
  const size_t woff = (size_t)wr * HID + kh * 8;

  v8f acc[4] = {};
#pragma unroll 2
  for (int kt = 0; kt < HID / 32; ++kt) {
    const int k0 = kt * 32;
    v16bf bhi = load_frag16(Whi + woff + k0, Whi + woff + k0 + 16);
    v16bf blo = load_frag16(Wlo + woff + k0, Wlo + woff + k0 + 16);
#pragma unroll
    for (int i = 0; i < 4; ++i) {
      v16bf ahi = load_frag16(Ahi + aoff[i] + k0, Ahi + aoff[i] + k0 + 16);
      v16bf alo = load_frag16(Alo + aoff[i] + k0, Alo + aoff[i] + k0 + 16);
      acc[i] = __builtin_amdgcn_wmma_f32_16x16x32_bf16(false, ahi, false, bhi,
                                                       (short)0, acc[i], false,
                                                       false);
      acc[i] = __builtin_amdgcn_wmma_f32_16x16x32_bf16(false, ahi, false, blo,
                                                       (short)0, acc[i], false,
                                                       false);
      acc[i] = __builtin_amdgcn_wmma_f32_16x16x32_bf16(false, alo, false, bhi,
                                                       (short)0, acc[i], false,
                                                       false);
    }
  }

  const int col = n0 + r;
  if (col < Ncols) {
    float bv = bias[col];
    const int mbase = m0 + kh * 8;
#pragma unroll
    for (int i = 0; i < 4; ++i) {
#pragma unroll
      for (int v = 0; v < 8; ++v) {
        // NT stores: don't let the 823MB logits stream evict out_W from L2.
        __builtin_nontemporal_store(
            acc[i][v] + bv,
            &Cout[(size_t)(mbase + i * 16 + v) * (size_t)Ncols + col]);
      }
    }
  }
}

// ---------------------------------------------------------------------------
// grow[r] = (Z[r,:].w2 + b2 >= 0)  (sigmoid monotone; clip brackets 0.5)
__global__ __launch_bounds__(256) void policy_kernel(
    const float* __restrict__ Z, const float* __restrict__ w2,
    const float* __restrict__ b2, float* __restrict__ grow, int R) {
  const int lane = threadIdx.x & 31;
  const int wave = threadIdx.x >> 5;
  const int row = blockIdx.x * 8 + wave;
  if (row >= R) return;
  const float* z = Z + (size_t)row * HID;
  float s = 0.0f;
#pragma unroll
  for (int i = 0; i < 8; ++i) s += z[lane + i * 32] * w2[lane + i * 32];
#pragma unroll
  for (int off = 16; off > 0; off >>= 1) s += __shfl_down(s, off, 32);
  if (lane == 0) grow[row] = (s + b2[0] >= 0.0f) ? 1.0f : 0.0f;
}

// C[(t*2+j)][h] = Base0[t][h] + SIB_SCALE*sib[j][h]
__global__ __launch_bounds__(256) void child_build(
    const float* __restrict__ Base0, const float* __restrict__ sib,
    float* __restrict__ Cc) {
  int idx = blockIdx.x * blockDim.x + threadIdx.x;
  int h = idx & (HID - 1);
  int rowc = idx >> 8;
  Cc[idx] = Base0[(rowc >> 1) * HID + h] + SIB_SCALE * sib[(rowc & 1) * HID + h];
}

// pooled -> split directly into bf16 hi/lo for the logits GEMM.
__global__ __launch_bounds__(256) void combine_kernel(
    const float* __restrict__ H0, const float* __restrict__ Base0,
    const float* __restrict__ Base1, const float* __restrict__ grow0,
    const float* __restrict__ grow1, const float* __restrict__ sib,
    ushort_t* __restrict__ phi, ushort_t* __restrict__ plo) {
  int idx = blockIdx.x * blockDim.x + threadIdx.x;  // 4096*256
  int h = idx & (HID - 1);
  int t = idx >> 8;
  float ss = SIB_SCALE * (sib[h] + sib[HID + h]);
  float g0 = grow0[t];
  float g10 = grow1[2 * t] * g0;
  float g11 = grow1[2 * t + 1] * g0;
  float s = H0[idx] + g0 * (2.0f * Base0[idx] + ss) +
            g10 * (2.0f * Base1[(size_t)(2 * t) * HID + h] + ss) +
            g11 * (2.0f * Base1[(size_t)(2 * t + 1) * HID + h] + ss);
  float cnt = 1.0f + 2.0f * g0 + 2.0f * (g10 + g11);
  float p = s / (cnt + 1e-8f);
  ushort_t hi, lo;
  split_bf16(p, hi, lo);
  phi[idx] = hi;
  plo[idx] = lo;
}

// out_W f32 -> bf16 hi/lo split.
__global__ __launch_bounds__(256) void splitW_kernel(
    const float* __restrict__ W, ushort_t* __restrict__ whi,
    ushort_t* __restrict__ wlo) {
  size_t idx = (size_t)blockIdx.x * blockDim.x + threadIdx.x;
  ushort_t hi, lo;
  split_bf16(W[idx], hi, lo);
  whi[idx] = hi;
  wlo[idx] = lo;
}

extern "C" void kernel_launch(void* const* d_in, const int* in_sizes, int n_in,
                              void* d_out, int out_size, void* d_ws,
                              size_t ws_size, hipStream_t stream) {
  const int* tokens = (const int*)d_in[0];
  const float* embedding = (const float*)d_in[1];
  const float* proj_W = (const float*)d_in[2];
  const float* proj_b = (const float*)d_in[3];
  const float* child_W = (const float*)d_in[4];
  const float* child_b = (const float*)d_in[5];
  const float* sib_emb = (const float*)d_in[6];
  const float* depth_emb = (const float*)d_in[7];
  const float* pol_W1 = (const float*)d_in[8];
  const float* pol_b1 = (const float*)d_in[9];
  const float* pol_w2 = (const float*)d_in[10];
  const float* pol_b2 = (const float*)d_in[11];
  const float* out_W = (const float*)d_in[12];
  const float* out_b = (const float*)d_in[13];
  float* logits = (float*)d_out;

  // Workspace layout.
  float* ws = (float*)d_ws;
  float* H0 = ws;                  // 4096*256 f32
  float* Base0 = H0 + (1u << 20);  // 4096*256 f32
  float* Cc = Base0 + (1u << 20);  // 8192*256 f32
  float* Base1 = Cc + (2u << 20);  // 8192*256 f32
  float* Z = Base1 + (2u << 20);   // 8192*256 f32
  float* grow0 = Z + (2u << 20);   // 4096
  float* grow1 = grow0 + 4096;     // 8192
  ushort_t* phi = (ushort_t*)(grow1 + 8192);  // 4096*256 bf16
  ushort_t* plo = phi + (1u << 20);
  ushort_t* whi = plo + (1u << 20);  // VOCAB*256 bf16
  ushort_t* wlo = whi + (size_t)VOCAB * HID;

  dim3 blk(256);
  dim3 gA(2, NTOK / 64);
  dim3 gA2(2, 8192 / 64);

  // 0) split out_W into bf16 hi/lo (independent of everything else)
  splitW_kernel<<<dim3(VOCAB), blk, 0, stream>>>(out_W, whi, wlo);
  // 1) H0 = gather(embedding)[tokens] @ proj_W^T + proj_b
  gemm_nt_wmma<0, true><<<gA, blk, 0, stream>>>(embedding, tokens, proj_W,
                                                proj_b, nullptr, H0, HID);
  // 2) Z0 = tanh(H0 @ pol_W1^T + pol_b1 + depth_emb[0]) ; grow0
  gemm_nt_wmma<1, false><<<gA, blk, 0, stream>>>(H0, nullptr, pol_W1, pol_b1,
                                                 depth_emb, Z, HID);
  policy_kernel<<<dim3(NTOK / 8), blk, 0, stream>>>(Z, pol_w2, pol_b2, grow0,
                                                    NTOK);
  // 3) Base0 = tanh(H0 @ child_W^T + child_b) ; children C
  gemm_nt_wmma<1, false><<<gA, blk, 0, stream>>>(H0, nullptr, child_W, child_b,
                                                 nullptr, Base0, HID);
  child_build<<<dim3((8192 * HID) / 256), blk, 0, stream>>>(Base0, sib_emb, Cc);
  // 4) Z1 = tanh(C @ pol_W1^T + pol_b1 + depth_emb[1]) ; grow1
  gemm_nt_wmma<1, false><<<gA2, blk, 0, stream>>>(Cc, nullptr, pol_W1, pol_b1,
                                                  depth_emb + HID, Z, HID);
  policy_kernel<<<dim3(8192 / 8), blk, 0, stream>>>(Z, pol_w2, pol_b2, grow1,
                                                    8192);
  // 5) Base1 = tanh(C @ child_W^T + child_b)
  gemm_nt_wmma<1, false><<<gA2, blk, 0, stream>>>(Cc, nullptr, child_W, child_b,
                                                  nullptr, Base1, HID);
  // 6) pooled -> bf16 hi/lo
  combine_kernel<<<dim3((NTOK * HID) / 256), blk, 0, stream>>>(
      H0, Base0, Base1, grow0, grow1, sib_emb, phi, plo);
  // 7) logits = pooled @ out_W^T + out_b  (bf16x3 split, f32 accumulate)
  int ntiles = (VOCAB + 15) / 16;  // 3142
  int gx = (ntiles + 7) / 8;       // 393
  gemm_bf16x3<<<dim3(gx, NTOK / 64), blk, 0, stream>>>(phi, plo, whi, wlo,
                                                       out_b, logits, VOCAB);
}
